// TransformerBlock_36661840839341
// MI455X (gfx1250) — matrix operand order repaired and checked by reading, not compiled, not run on hardware
//
#include <hip/hip_runtime.h>
#include <hip/hip_bf16.h>
#include <math.h>

#define BB 8
#define NN 2048
#define DP 64
#define DM 128
#define KNB 32
#define NPTS (BB * NN)          // 16384

typedef __attribute__((ext_vector_type(16))) _Float16 v16h;
typedef __attribute__((ext_vector_type(8)))  _Float16 v8h;
typedef __attribute__((ext_vector_type(8)))  float    v8f;

// ---------------------------------------------------------------------------
// WMMA helpers (gfx1250, wave32, 16x16x32 f16 -> f32)
// ---------------------------------------------------------------------------
__device__ inline v8f wmma_f16(v16h a, v16h b, v8f c) {
    return __builtin_amdgcn_wmma_f32_16x16x32_f16(
        /*neg_a=*/false, a, /*neg_b=*/false, b,
        /*c_mod=*/(short)0, c, /*reuse_a=*/false, /*reuse_b=*/false);
}

// A fragment (16x32, 16-bit, row-major source with leading dim lda halves).
// ISA layout: lane m=lane&15, khalf=lane>>4; VGPR0-3 hold K=khalf*8+0..7,
// VGPR4-7 hold K=khalf*8+16..23  -> two contiguous 8-half (16B) loads.
__device__ inline v16h load_a_frag(const _Float16* __restrict__ A, int lda, int lane) {
    const _Float16* p = A + (size_t)(lane & 15) * lda + ((lane >> 4) << 3);
    v8h lo = *(const v8h*)(p);
    v8h hi = *(const v8h*)(p + 16);
    v16h r;
#pragma unroll
    for (int i = 0; i < 8; ++i) { r[i] = lo[i]; r[8 + i] = hi[i]; }
    return r;
}

// B fragment (32x16, 16-bit) from an N-major (transposed) weight matrix:
// Bt[n][k], leading dim ldb = K. Lane n=lane&15 holds K=(lane>>4)*8 + 0..7
// and K=(lane>>4)*8 + 16..23 -> two contiguous 8-half (16B) loads.
__device__ inline v16h load_b_frag(const _Float16* __restrict__ Bt, int ldb, int lane) {
    const _Float16* p = Bt + (size_t)(lane & 15) * ldb + ((lane >> 4) << 3);
    v8h lo = *(const v8h*)(p);
    v8h hi = *(const v8h*)(p + 16);
    v16h r;
#pragma unroll
    for (int i = 0; i < 8; ++i) { r[i] = lo[i]; r[8 + i] = hi[i]; }
    return r;
}

// ---------------------------------------------------------------------------
// Weight conversion: f32 (R x C) -> f16 transposed (C x R)
// ---------------------------------------------------------------------------
__global__ void k_cvt_t(const float* __restrict__ W, _Float16* __restrict__ Wt,
                        int R, int C) {
    int i = blockIdx.x * blockDim.x + threadIdx.x;
    if (i >= R * C) return;
    int r = i / C, c = i % C;
    Wt[(size_t)c * R + r] = (_Float16)W[(size_t)r * C + c];
}

__global__ void k_cvt(const float* __restrict__ A, _Float16* __restrict__ Ah, int n) {
    int i = blockIdx.x * blockDim.x + threadIdx.x;
    if (i < n) Ah[i] = (_Float16)A[i];
}

// ---------------------------------------------------------------------------
// kNN: one thread per query point, candidates tiled through LDS,
// running top-32 by replace-worst, final insertion sort (ascending).
// ---------------------------------------------------------------------------
__global__ __launch_bounds__(256) void k_knn(const float* __restrict__ xyz,
                                             int* __restrict__ knn) {
    __shared__ float sx[256], sy[256], sz[256];
    int pt = blockIdx.x * 256 + threadIdx.x;      // 0..16383
    int b  = pt >> 11;
    float qx = xyz[(size_t)pt * 3 + 0];
    float qy = xyz[(size_t)pt * 3 + 1];
    float qz = xyz[(size_t)pt * 3 + 2];

    float bd[KNB]; int bi[KNB];
#pragma unroll
    for (int i = 0; i < KNB; ++i) { bd[i] = 3.4e38f; bi[i] = 0; }
    float wd = 3.4e38f; int wp = 0;

    for (int t = 0; t < NN; t += 256) {
        __syncthreads();
        int j = t + threadIdx.x;
        size_t base = ((size_t)b * NN + j) * 3;
        sx[threadIdx.x] = xyz[base + 0];
        sy[threadIdx.x] = xyz[base + 1];
        sz[threadIdx.x] = xyz[base + 2];
        __syncthreads();
        for (int jj = 0; jj < 256; ++jj) {
            float dx = qx - sx[jj], dy = qy - sy[jj], dz = qz - sz[jj];
            float d = dx * dx + dy * dy + dz * dz;
            if (d < wd) {
                bd[wp] = d; bi[wp] = t + jj;
                wd = bd[0]; wp = 0;
                for (int i = 1; i < KNB; ++i)
                    if (bd[i] > wd) { wd = bd[i]; wp = i; }
            }
        }
    }
    // ascending insertion sort (matches argsort order of the reference)
    for (int i = 1; i < KNB; ++i) {
        float dv = bd[i]; int iv = bi[i]; int j = i - 1;
        while (j >= 0 && bd[j] > dv) { bd[j + 1] = bd[j]; bi[j + 1] = bi[j]; --j; }
        bd[j + 1] = dv; bi[j + 1] = iv;
    }
    for (int i = 0; i < KNB; ++i) knn[(size_t)pt * KNB + i] = bi[i];
}

// ---------------------------------------------------------------------------
// x = features @ w1 + b1   (M=16384, K=64, N=128) -> f16
// ---------------------------------------------------------------------------
__global__ __launch_bounds__(128) void k_fc1(const _Float16* __restrict__ A,
                                             const _Float16* __restrict__ Bt,
                                             const float* __restrict__ bias,
                                             _Float16* __restrict__ X) {
    int lane = threadIdx.x & 31, wv = threadIdx.x >> 5;
    int m0 = (blockIdx.x * 4 + wv) * 16;
    int mrow = (lane >> 4) << 3, ncol = lane & 15;
    v16h a0 = load_a_frag(A + (size_t)m0 * DP +  0, DP, lane);
    v16h a1 = load_a_frag(A + (size_t)m0 * DP + 32, DP, lane);
#pragma unroll
    for (int nt = 0; nt < 8; ++nt) {
        v8f acc = {};
        acc = wmma_f16(a0, load_b_frag(Bt + (size_t)(nt * 16) * DP +  0, DP, lane), acc);
        acc = wmma_f16(a1, load_b_frag(Bt + (size_t)(nt * 16) * DP + 32, DP, lane), acc);
        float bs = bias[nt * 16 + ncol];
#pragma unroll
        for (int r = 0; r < 8; ++r)
            X[(size_t)(m0 + mrow + r) * DM + nt * 16 + ncol] = (_Float16)(acc[r] + bs);
    }
}

// ---------------------------------------------------------------------------
// q/k/v = x @ {wq,wk,wv}  (M=16384, K=128, N=128) -> f32
// ---------------------------------------------------------------------------
__global__ __launch_bounds__(128) void k_qkv(const _Float16* __restrict__ X,
                                             const _Float16* __restrict__ Wq,
                                             const _Float16* __restrict__ Wk,
                                             const _Float16* __restrict__ Wv,
                                             float* __restrict__ q,
                                             float* __restrict__ k,
                                             float* __restrict__ v) {
    int lane = threadIdx.x & 31, wv = threadIdx.x >> 5;
    int m0 = (blockIdx.x * 4 + wv) * 16;
    int mrow = (lane >> 4) << 3, ncol = lane & 15;
    v16h a[4];
#pragma unroll
    for (int i = 0; i < 4; ++i)
        a[i] = load_a_frag(X + (size_t)m0 * DM + 32 * i, DM, lane);
    const _Float16* Ws[3] = {Wq, Wk, Wv};
    float* outs[3] = {q, k, v};
#pragma unroll
    for (int w = 0; w < 3; ++w) {
        const _Float16* W = Ws[w];
        float* O = outs[w];
#pragma unroll
        for (int nt = 0; nt < 8; ++nt) {
            v8f acc = {};
#pragma unroll
            for (int i = 0; i < 4; ++i)
                acc = wmma_f16(a[i], load_b_frag(W + (size_t)(nt * 16) * DM + 32 * i, DM, lane), acc);
#pragma unroll
            for (int r = 0; r < 8; ++r)
                O[(size_t)(m0 + mrow + r) * DM + nt * 16 + ncol] = acc[r];
        }
    }
}

// ---------------------------------------------------------------------------
// Fused attention kernel: one wave per point (32 neighbor rows x 128 cols).
// pos = relu(rel@dw1+db1)@dw2+db2 ; attn_in = q - k_nb + pos ;
// logits = relu(attn_in@gw1+gb1)@gw2+gb2 ; softmax over 32 neighbors ;
// attn -> d_out, res = sum_k attn*(v_nb+pos) -> res16.
// ---------------------------------------------------------------------------
__global__ __launch_bounds__(128) void k_attn(
    const float* __restrict__ xyz, const int* __restrict__ knn,
    const float* __restrict__ q, const float* __restrict__ kmat,
    const float* __restrict__ vmat,
    const float* __restrict__ dw1, const float* __restrict__ db1,
    const _Float16* __restrict__ dw2t, const float* __restrict__ db2,
    const _Float16* __restrict__ gw1t, const float* __restrict__ gb1,
    const _Float16* __restrict__ gw2t, const float* __restrict__ gb2,
    float* __restrict__ attn_out, _Float16* __restrict__ res16) {

    __shared__ __align__(32) _Float16 abuf[4][KNB][DM];   // 32 KB
    __shared__ __align__(32) _Float16 posbuf[4][KNB][DM]; // 32 KB
    __shared__ int ibuf[4][KNB];

    int lane = threadIdx.x & 31, wv = threadIdx.x >> 5;
    int pt = blockIdx.x * 4 + wv;
    int b  = pt >> 11;
    int mrow = (lane >> 4) << 3, ncol = lane & 15;

    // Stage 1: neighbor indices, rel, first pos-enc layer (K=3, scalar)
    {
        int row = lane;
        int idx = knn[(size_t)pt * KNB + row];
        ibuf[wv][row] = idx;
        size_t nb = ((size_t)b * NN + idx) * 3;
        float rx = xyz[(size_t)pt * 3 + 0] - xyz[nb + 0];
        float ry = xyz[(size_t)pt * 3 + 1] - xyz[nb + 1];
        float rz = xyz[(size_t)pt * 3 + 2] - xyz[nb + 2];
        for (int c = 0; c < DM; ++c) {
            float h = db1[c] + rx * dw1[c] + ry * dw1[DM + c] + rz * dw1[2 * DM + c];
            abuf[wv][row][c] = (_Float16)fmaxf(h, 0.0f);
        }
    }
    __syncthreads();

    const float* qrow = q + (size_t)pt * DM;
    v8f logit[2][8];

    // Stage 2: pos = h1 @ dw2t + db2; build attn_in in abuf, pos in posbuf
#pragma unroll
    for (int mh = 0; mh < 2; ++mh) {
        v16h a[4];
#pragma unroll
        for (int i = 0; i < 4; ++i)
            a[i] = load_a_frag(&abuf[wv][mh * 16][0] + 32 * i, DM, lane);
#pragma unroll
        for (int nt = 0; nt < 8; ++nt) {
            v8f acc = {};
#pragma unroll
            for (int i = 0; i < 4; ++i)
                acc = wmma_f16(a[i], load_b_frag(dw2t + (size_t)(nt * 16) * DM + 32 * i, DM, lane), acc);
            float bsv = db2[nt * 16 + ncol];
            float qv  = qrow[nt * 16 + ncol];
#pragma unroll
            for (int r = 0; r < 8; ++r) {
                int row = mh * 16 + mrow + r;
                float pos = acc[r] + bsv;
                posbuf[wv][row][nt * 16 + ncol] = (_Float16)pos;
                int idx = ibuf[wv][row];
                float kv = kmat[((size_t)b * NN + idx) * DM + nt * 16 + ncol];
                abuf[wv][row][nt * 16 + ncol] = (_Float16)(qv - kv + pos);
            }
        }
    }
    __syncthreads();

    // Stage 3: logits = relu(attn_in@gw1+gb1)@gw2+gb2
#pragma unroll
    for (int mh = 0; mh < 2; ++mh) {
        v16h a[4];
#pragma unroll
        for (int i = 0; i < 4; ++i)
            a[i] = load_a_frag(&abuf[wv][mh * 16][0] + 32 * i, DM, lane);
#pragma unroll
        for (int nt = 0; nt < 8; ++nt) {
            v8f acc = {};
#pragma unroll
            for (int i = 0; i < 4; ++i)
                acc = wmma_f16(a[i], load_b_frag(gw1t + (size_t)(nt * 16) * DM + 32 * i, DM, lane), acc);
            float bb = gb1[nt * 16 + ncol];
#pragma unroll
            for (int r = 0; r < 8; ++r)
                abuf[wv][mh * 16 + mrow + r][nt * 16 + ncol] =
                    (_Float16)fmaxf(acc[r] + bb, 0.0f);
        }
        v16h h[4];
#pragma unroll
        for (int i = 0; i < 4; ++i)
            h[i] = load_a_frag(&abuf[wv][mh * 16][0] + 32 * i, DM, lane);
#pragma unroll
        for (int nt = 0; nt < 8; ++nt) {
            v8f acc = {};
#pragma unroll
            for (int i = 0; i < 4; ++i)
                acc = wmma_f16(h[i], load_b_frag(gw2t + (size_t)(nt * 16) * DM + 32 * i, DM, lane), acc);
            float bb = gb2[nt * 16 + ncol];
#pragma unroll
            for (int r = 0; r < 8; ++r) acc[r] += bb;
            logit[mh][nt] = acc;
        }
    }
    __syncthreads();

    // Stage 4: softmax over 32 neighbors (lane p <-> p^16 hold the two halves
    // of each column in the C-layout), attn out, weighted sum res.
    const float rs = 0.17677669529663687f;  // 1/sqrt(HEAD_DIM=32)
#pragma unroll
    for (int nt = 0; nt < 8; ++nt) {
        int c = nt * 16 + ncol;
        float mx = -3.4e38f;
#pragma unroll
        for (int mh = 0; mh < 2; ++mh)
#pragma unroll
            for (int r = 0; r < 8; ++r) mx = fmaxf(mx, logit[mh][nt][r]);
        mx = fmaxf(mx, __shfl_xor(mx, 16, 32));
        float e[2][8];
        float sum = 0.0f;
#pragma unroll
        for (int mh = 0; mh < 2; ++mh)
#pragma unroll
            for (int r = 0; r < 8; ++r) {
                e[mh][r] = __expf((logit[mh][nt][r] - mx) * rs);
                sum += e[mh][r];
            }
        sum += __shfl_xor(sum, 16, 32);
        float inv = 1.0f / sum;
        float racc = 0.0f;
#pragma unroll
        for (int mh = 0; mh < 2; ++mh)
#pragma unroll
            for (int r = 0; r < 8; ++r) {
                int row = mh * 16 + mrow + r;
                float aval = e[mh][r] * inv;
                attn_out[(((size_t)pt * KNB) + row) * DM + c] = aval;
                int idx = ibuf[wv][row];
                float vv = vmat[((size_t)b * NN + idx) * DM + c];
                float pv = (float)posbuf[wv][row][c];
                racc += aval * (vv + pv);
            }
        racc += __shfl_xor(racc, 16, 32);
        if (lane < 16) res16[(size_t)pt * DM + c] = (_Float16)racc;
    }
}

// ---------------------------------------------------------------------------
// out = (res @ wo) @ w2 + b2 + features   (two chained WMMA GEMMs via LDS)
// ---------------------------------------------------------------------------
__global__ __launch_bounds__(128) void k_out(const _Float16* __restrict__ R,
                                             const _Float16* __restrict__ Wot,
                                             const _Float16* __restrict__ W2t,
                                             const float* __restrict__ b2,
                                             const float* __restrict__ features,
                                             float* __restrict__ out) {
    __shared__ __align__(32) _Float16 tbuf[4][16][DM];
    int lane = threadIdx.x & 31, wv = threadIdx.x >> 5;
    int m0 = (blockIdx.x * 4 + wv) * 16;
    int mrow = (lane >> 4) << 3, ncol = lane & 15;
    v16h a[4];
#pragma unroll
    for (int i = 0; i < 4; ++i)
        a[i] = load_a_frag(R + (size_t)m0 * DM + 32 * i, DM, lane);
#pragma unroll
    for (int nt = 0; nt < 8; ++nt) {
        v8f acc = {};
#pragma unroll
        for (int i = 0; i < 4; ++i)
            acc = wmma_f16(a[i], load_b_frag(Wot + (size_t)(nt * 16) * DM + 32 * i, DM, lane), acc);
#pragma unroll
        for (int r = 0; r < 8; ++r)
            tbuf[wv][mrow + r][nt * 16 + ncol] = (_Float16)acc[r];
    }
    v16h t[4];
#pragma unroll
    for (int i = 0; i < 4; ++i)
        t[i] = load_a_frag(&tbuf[wv][0][0] + 32 * i, DM, lane);
#pragma unroll
    for (int nt = 0; nt < 4; ++nt) {   // N = 64
        v8f acc = {};
#pragma unroll
        for (int i = 0; i < 4; ++i)
            acc = wmma_f16(t[i], load_b_frag(W2t + (size_t)(nt * 16) * DM + 32 * i, DM, lane), acc);
#pragma unroll
        for (int r = 0; r < 8; ++r) {
            int m = m0 + mrow + r, c = nt * 16 + ncol;
            out[(size_t)m * DP + c] = acc[r] + b2[c] + features[(size_t)m * DP + c];
        }
    }
}

// ---------------------------------------------------------------------------
// Host launcher
// ---------------------------------------------------------------------------
extern "C" void kernel_launch(void* const* d_in, const int* in_sizes, int n_in,
                              void* d_out, int out_size, void* d_ws, size_t ws_size,
                              hipStream_t stream) {
    const float* xyz      = (const float*)d_in[0];
    const float* features = (const float*)d_in[1];
    const float* w1  = (const float*)d_in[2];
    const float* b1  = (const float*)d_in[3];
    const float* w2  = (const float*)d_in[4];
    const float* b2  = (const float*)d_in[5];
    const float* dw1 = (const float*)d_in[6];
    const float* db1 = (const float*)d_in[7];
    const float* dw2 = (const float*)d_in[8];
    const float* db2 = (const float*)d_in[9];
    const float* gw1 = (const float*)d_in[10];
    const float* gb1 = (const float*)d_in[11];
    const float* gw2 = (const float*)d_in[12];
    const float* gb2 = (const float*)d_in[13];
    const float* wq  = (const float*)d_in[14];
    const float* wk  = (const float*)d_in[15];
    const float* wv  = (const float*)d_in[16];
    const float* wo  = (const float*)d_in[17];

    float* out_ptr  = (float*)d_out;                       // (16384, 64)
    float* attn_ptr = (float*)d_out + (size_t)NPTS * DP;   // (16384, 32, 128)

    // workspace carve-up (256B aligned)
    char* base = (char*)d_ws;
    size_t off = 0;
    auto alloc = [&](size_t bytes) -> char* {
        char* p = base + off;
        off += (bytes + 255) & ~(size_t)255;
        return p;
    };
    _Float16* w1t  = (_Float16*)alloc((size_t)DP * DM * 2);
    _Float16* wqt  = (_Float16*)alloc((size_t)DM * DM * 2);
    _Float16* wkt  = (_Float16*)alloc((size_t)DM * DM * 2);
    _Float16* wvt  = (_Float16*)alloc((size_t)DM * DM * 2);
    _Float16* wot  = (_Float16*)alloc((size_t)DM * DM * 2);
    _Float16* dw2t = (_Float16*)alloc((size_t)DM * DM * 2);
    _Float16* gw1t = (_Float16*)alloc((size_t)DM * DM * 2);
    _Float16* gw2t = (_Float16*)alloc((size_t)DM * DM * 2);
    _Float16* w2t  = (_Float16*)alloc((size_t)DM * DP * 2);
    int*      knn  = (int*)     alloc((size_t)NPTS * KNB * 4);
    _Float16* f16  = (_Float16*)alloc((size_t)NPTS * DP * 2);
    _Float16* x16  = (_Float16*)alloc((size_t)NPTS * DM * 2);
    float*    qb   = (float*)   alloc((size_t)NPTS * DM * 4);
    float*    kb   = (float*)   alloc((size_t)NPTS * DM * 4);
    float*    vb   = (float*)   alloc((size_t)NPTS * DM * 4);
    _Float16* r16  = (_Float16*)alloc((size_t)NPTS * DM * 2);
    (void)ws_size; (void)n_in; (void)in_sizes; (void)out_size;

    // weight conversion (+ transpose to N-major f16)
    k_cvt_t<<<(DP * DM + 255) / 256, 256, 0, stream>>>(w1,  w1t,  DP, DM);
    k_cvt_t<<<(DM * DM + 255) / 256, 256, 0, stream>>>(wq,  wqt,  DM, DM);
    k_cvt_t<<<(DM * DM + 255) / 256, 256, 0, stream>>>(wk,  wkt,  DM, DM);
    k_cvt_t<<<(DM * DM + 255) / 256, 256, 0, stream>>>(wv,  wvt,  DM, DM);
    k_cvt_t<<<(DM * DM + 255) / 256, 256, 0, stream>>>(wo,  wot,  DM, DM);
    k_cvt_t<<<(DM * DM + 255) / 256, 256, 0, stream>>>(dw2, dw2t, DM, DM);
    k_cvt_t<<<(DM * DM + 255) / 256, 256, 0, stream>>>(gw1, gw1t, DM, DM);
    k_cvt_t<<<(DM * DM + 255) / 256, 256, 0, stream>>>(gw2, gw2t, DM, DM);
    k_cvt_t<<<(DM * DP + 255) / 256, 256, 0, stream>>>(w2,  w2t,  DM, DP);
    k_cvt<<<(NPTS * DP + 255) / 256, 256, 0, stream>>>(features, f16, NPTS * DP);

    // kNN top-32
    k_knn<<<NPTS / 256, 256, 0, stream>>>(xyz, knn);

    // projections
    k_fc1<<<NPTS / 16 / 4, 128, 0, stream>>>(f16, w1t, b1, x16);
    k_qkv<<<NPTS / 16 / 4, 128, 0, stream>>>(x16, wqt, wkt, wvt, qb, kb, vb);

    // fused attention (one wave per point)
    k_attn<<<NPTS / 4, 128, 0, stream>>>(xyz, knn, qb, kb, vb,
                                         dw1, db1, dw2t, db2,
                                         gw1t, gb1, gw2t, gb2,
                                         attn_ptr, r16);

    // output projection + residual
    k_out<<<NPTS / 16 / 4, 128, 0, stream>>>(r16, wot, w2t, b2, features, out_ptr);
}
